// Router_15942918603252
// MI455X (gfx1250) — matrix operand order repaired and checked by reading, not compiled, not run on hardware
//
#include <hip/hip_runtime.h>

typedef __bf16 bf16_t;
typedef __attribute__((ext_vector_type(16))) __bf16 v16bf;
typedef __attribute__((ext_vector_type(8)))  __bf16 v8bf;
typedef __attribute__((ext_vector_type(8)))  float  v8f;
typedef __attribute__((ext_vector_type(4)))  int    v4i;

namespace {
constexpr int Bn   = 16384;
constexpr int INn  = 512;
constexpr int Hn   = 1024;
constexpr int OUTn = 512;
constexpr int Mn   = 8;
constexpr int Rn   = 256;
constexpr int CYC  = 4;
constexpr int BK   = 64;   // K-chunk depth
constexpr int LDSS = 72;   // 64 + 8 pad bf16 -> 144B rows (16B aligned)
}

// ---- async global->LDS staging (gfx1250) ---------------------------------
#if defined(__has_builtin)
#  if __has_builtin(__builtin_amdgcn_global_load_async_to_lds_b128)
#    define HAVE_ASYNC_LDS 1
#  endif
#endif
#ifndef HAVE_ASYNC_LDS
#  define HAVE_ASYNC_LDS 0
#endif

__device__ __forceinline__ void async_copy16(bf16_t* lds_dst, const bf16_t* gsrc) {
#if HAVE_ASYNC_LDS
  __builtin_amdgcn_global_load_async_to_lds_b128(
      (__attribute__((address_space(1))) v4i*)(void*)gsrc,
      (__attribute__((address_space(3))) v4i*)(void*)lds_dst, 0, 0);
#else
  *(uint4*)lds_dst = *(const uint4*)gsrc;
#endif
}

__device__ __forceinline__ void wait_async0() {
#if HAVE_ASYNC_LDS
#  if __has_builtin(__builtin_amdgcn_s_wait_asynccnt)
  __builtin_amdgcn_s_wait_asynccnt(0);
#  else
  asm volatile("s_wait_asynccnt 0x0" ::: "memory");
#  endif
#endif
}

// ---- fragment loads from LDS (K-contiguous rows) -------------------------
// A-frag (16x32): lane l -> row l%16; elems 0..7 = K[8*(l/16)+0..7],
//                 elems 8..15 = K[16+8*(l/16)+0..7]
__device__ __forceinline__ v16bf frag_a(const bf16_t* row, int half) {
  const bf16_t* p = row + half * 8;
  v8bf lo = *(const v8bf*)(p);
  v8bf hi = *(const v8bf*)(p + 16);
  v16bf r;
#pragma unroll
  for (int i = 0; i < 8; ++i) { r[i] = lo[i]; r[i + 8] = hi[i]; }
  return r;
}
// B-frag (32x16): lane l -> col l%16; elems 0..7 = K[8*(l/16)+0..7],
//                 elems 8..15 = K[16+8*(l/16)+0..7]
__device__ __forceinline__ v16bf frag_b(const bf16_t* row, int half) {
  const bf16_t* p = row + half * 8;
  v8bf lo = *(const v8bf*)(p);
  v8bf hi = *(const v8bf*)(p + 16);
  v16bf r;
#pragma unroll
  for (int i = 0; i < 8; ++i) { r[i] = lo[i]; r[i + 8] = hi[i]; }
  return r;
}

__device__ __forceinline__ v8f wmma_bf16(v16bf a, v16bf b, v8f c) {
  return __builtin_amdgcn_wmma_f32_16x16x32_bf16(false, a, false, b, (short)0, c,
                                                 false, false);
}

// ---- small conversion kernels -------------------------------------------
__global__ void conv_f32_to_bf16(const float* __restrict__ X, bf16_t* __restrict__ Y) {
  size_t i = (size_t)blockIdx.x * 256 + threadIdx.x;
  Y[i] = (bf16_t)X[i];
}

// Wt[n*K + k] = W[k*N + n], batched over blockIdx.y
__global__ void convt_f32_to_bf16t(const float* __restrict__ W, bf16_t* __restrict__ Wt,
                                   int K, int N) {
  size_t base = (size_t)blockIdx.y * K * N;
  int idx = blockIdx.x * 256 + threadIdx.x;
  int k = idx / N, n = idx % N;
  Wt[base + (size_t)n * K + k] = (bf16_t)W[base + (size_t)k * N + n];
}

// xm = orig * (c*Wc + bc), bf16 out
__global__ void modulate_kernel(const bf16_t* __restrict__ orig, const float* __restrict__ Wc,
                                const float* __restrict__ bc, bf16_t* __restrict__ xm,
                                float cyc, int H) {
  size_t i = (size_t)blockIdx.x * 256 + threadIdx.x;
  int h = (int)(i % H);
  float proj = cyc * Wc[h] + bc[h];
  xm[i] = (bf16_t)((float)orig[i] * proj);
}

// probs = softmax(r @ Wr2 + br2); one thread per row (K=256, M=8)
__global__ void router_kernel(const float* __restrict__ r, const float* __restrict__ Wr2,
                              const float* __restrict__ br2, float* __restrict__ probs) {
  int row = blockIdx.x * 256 + threadIdx.x;
  const float* rr = r + (size_t)row * Rn;
  float acc[Mn];
#pragma unroll
  for (int m = 0; m < Mn; ++m) acc[m] = br2[m];
  for (int k = 0; k < Rn; ++k) {
    float v = rr[k];
#pragma unroll
    for (int m = 0; m < Mn; ++m) acc[m] = fmaf(v, Wr2[k * Mn + m], acc[m]);
  }
  float mx = acc[0];
#pragma unroll
  for (int m = 1; m < Mn; ++m) mx = fmaxf(mx, acc[m]);
  float s = 0.f;
#pragma unroll
  for (int m = 0; m < Mn; ++m) { acc[m] = __expf(acc[m] - mx); s += acc[m]; }
  float inv = 1.f / s;
#pragma unroll
  for (int m = 0; m < Mn; ++m) probs[(size_t)row * Mn + m] = acc[m] * inv;
}

// ---- generic WMMA GEMM: C[M,N] = A[M,K](bf16) * Bt[N,K](bf16)^T + bias ----
// 256 thr (8 waves), block tile 128x128, BK=64, double-buffered async LDS.
template <bool RELU, bool OUT_BF16>
__global__ __launch_bounds__(256) void gemm_bf16_kernel(
    const bf16_t* __restrict__ A, const bf16_t* __restrict__ Bt,
    const float* __restrict__ bias, void* __restrict__ Cout, int Ntot, int K) {
  const int tid = threadIdx.x;
  const int wid = tid >> 5, lane = tid & 31;
  const int wm = wid >> 2, wn = wid & 3;         // 2x4 wave grid
  const int row0 = blockIdx.y * 128, col0 = blockIdx.x * 128;
  const int fr = lane & 15, half = lane >> 4;

  __shared__ alignas(16) bf16_t As[2][128][LDSS];
  __shared__ alignas(16) bf16_t Bs[2][128][LDSS];

  v8f acc[4][2];
#pragma unroll
  for (int a = 0; a < 4; ++a)
#pragma unroll
    for (int b = 0; b < 2; ++b) acc[a][b] = (v8f){0, 0, 0, 0, 0, 0, 0, 0};

  const int sr = tid >> 1;            // staging row 0..127
  const int sk = (tid & 1) * 32;      // element offset 0 / 32 within 64
  const int nch = K / BK;

  auto issue = [&](int c) {
    const int k0 = c * BK;
    const bf16_t* ga = A  + (size_t)(row0 + sr) * K + k0 + sk;
    const bf16_t* gb = Bt + (size_t)(col0 + sr) * K + k0 + sk;
    bf16_t* la = &As[c & 1][sr][sk];
    bf16_t* lb = &Bs[c & 1][sr][sk];
#pragma unroll
    for (int v = 0; v < 4; ++v) {
      async_copy16(la + v * 8, ga + v * 8);
      async_copy16(lb + v * 8, gb + v * 8);
    }
  };

  issue(0);
  for (int c = 0; c < nch; ++c) {
    wait_async0();
    __syncthreads();                  // chunk c visible; all waves past chunk c-1
    if (c + 1 < nch) issue(c + 1);    // DMA next chunk while computing this one
    const int b = c & 1;
#pragma unroll
    for (int kh = 0; kh < 2; ++kh) {
      v16bf bf0 = frag_b(&Bs[b][wn * 32 + fr][kh * 32], half);
      v16bf bf1 = frag_b(&Bs[b][wn * 32 + 16 + fr][kh * 32], half);
#pragma unroll
      for (int tm = 0; tm < 4; ++tm) {
        v16bf af = frag_a(&As[b][wm * 64 + tm * 16 + fr][kh * 32], half);
        acc[tm][0] = wmma_bf16(af, bf0, acc[tm][0]);
        acc[tm][1] = wmma_bf16(af, bf1, acc[tm][1]);
      }
    }
  }
  // epilogue: C VGPR i -> row = i + 8*half, col = lane&15
#pragma unroll
  for (int tn = 0; tn < 2; ++tn) {
    int cg = col0 + wn * 32 + tn * 16 + fr;
    float bv = bias[cg];
#pragma unroll
    for (int tm = 0; tm < 4; ++tm) {
      int rg0 = row0 + wm * 64 + tm * 16 + half * 8;
#pragma unroll
      for (int i = 0; i < 8; ++i) {
        float v = acc[tm][tn][i] + bv;
        if (RELU) v = fmaxf(v, 0.f);
        size_t idx = (size_t)(rg0 + i) * Ntot + cg;
        if (OUT_BF16) ((bf16_t*)Cout)[idx] = (bf16_t)v;
        else          ((float*)Cout)[idx]  = v;
      }
    }
  }
}

// ---- fused MoE: h_new = sum_m probs[:,m] * relu(h @ Wmod[m] + b_mod[m]) ----
// Flattened m*kchunk pipeline so double-buffering streams across experts.
__global__ __launch_bounds__(256) void expert_moe_kernel(
    const bf16_t* __restrict__ Hc, const bf16_t* __restrict__ WtMod, // [M][N=H][K=H]
    const float* __restrict__ b_mod,                                 // [M][H]
    const float* __restrict__ probs,                                 // [B][M]
    bf16_t* __restrict__ Hnxt, int Hd) {
  const int tid = threadIdx.x;
  const int wid = tid >> 5, lane = tid & 31;
  const int wm = wid >> 2, wn = wid & 3;
  const int row0 = blockIdx.y * 128, col0 = blockIdx.x * 128;
  const int fr = lane & 15, half = lane >> 4;

  __shared__ alignas(16) bf16_t As[2][128][LDSS];
  __shared__ alignas(16) bf16_t Bs[2][128][LDSS];

  v8f accf[4][2], acc[4][2];
#pragma unroll
  for (int a = 0; a < 4; ++a)
#pragma unroll
    for (int b = 0; b < 2; ++b) {
      accf[a][b] = (v8f){0, 0, 0, 0, 0, 0, 0, 0};
      acc[a][b]  = (v8f){0, 0, 0, 0, 0, 0, 0, 0};
    }

  const int sr = tid >> 1;
  const int sk = (tid & 1) * 32;
  const int kpm = Hd / BK;            // chunks per expert (16)
  const int nch = kpm * Mn;           // 128

  auto issue = [&](int c) {
    const int m = c / kpm;
    const int k0 = (c % kpm) * BK;
    const bf16_t* ga = Hc + (size_t)(row0 + sr) * Hd + k0 + sk;
    const bf16_t* gb = WtMod + (size_t)m * Hd * Hd + (size_t)(col0 + sr) * Hd + k0 + sk;
    bf16_t* la = &As[c & 1][sr][sk];
    bf16_t* lb = &Bs[c & 1][sr][sk];
#pragma unroll
    for (int v = 0; v < 4; ++v) {
      async_copy16(la + v * 8, ga + v * 8);
      async_copy16(lb + v * 8, gb + v * 8);
    }
  };

  issue(0);
  for (int c = 0; c < nch; ++c) {
    wait_async0();
    __syncthreads();
    if (c + 1 < nch) issue(c + 1);
    const int b = c & 1;
#pragma unroll
    for (int kh = 0; kh < 2; ++kh) {
      v16bf bf0 = frag_b(&Bs[b][wn * 32 + fr][kh * 32], half);
      v16bf bf1 = frag_b(&Bs[b][wn * 32 + 16 + fr][kh * 32], half);
#pragma unroll
      for (int tm = 0; tm < 4; ++tm) {
        v16bf af = frag_a(&As[b][wm * 64 + tm * 16 + fr][kh * 32], half);
        acc[tm][0] = wmma_bf16(af, bf0, acc[tm][0]);
        acc[tm][1] = wmma_bf16(af, bf1, acc[tm][1]);
      }
    }
    if ((c % kpm) == kpm - 1) {
      // expert m finished: relu(+bias), weight by probs[row,m], accumulate
      const int m = c / kpm;
      float bv0 = b_mod[m * Hd + col0 + wn * 32 + fr];
      float bv1 = b_mod[m * Hd + col0 + wn * 32 + 16 + fr];
#pragma unroll
      for (int tm = 0; tm < 4; ++tm) {
        int rg0 = row0 + wm * 64 + tm * 16 + half * 8;
#pragma unroll
        for (int i = 0; i < 8; ++i) {
          float p = probs[(size_t)(rg0 + i) * Mn + m];
          accf[tm][0][i] += p * fmaxf(acc[tm][0][i] + bv0, 0.f);
          accf[tm][1][i] += p * fmaxf(acc[tm][1][i] + bv1, 0.f);
          acc[tm][0][i] = 0.f;
          acc[tm][1][i] = 0.f;
        }
      }
    }
  }
  // store h_new (bf16)
#pragma unroll
  for (int tn = 0; tn < 2; ++tn) {
    int cg = col0 + wn * 32 + tn * 16 + fr;
#pragma unroll
    for (int tm = 0; tm < 4; ++tm) {
      int rg0 = row0 + wm * 64 + tm * 16 + half * 8;
#pragma unroll
      for (int i = 0; i < 8; ++i)
        Hnxt[(size_t)(rg0 + i) * Hd + cg] = (bf16_t)accf[tm][tn][i];
    }
  }
}

// ---- host driver ---------------------------------------------------------
extern "C" void kernel_launch(void* const* d_in, const int* in_sizes, int n_in,
                              void* d_out, int out_size, void* d_ws, size_t ws_size,
                              hipStream_t stream) {
  (void)in_sizes; (void)n_in; (void)out_size; (void)ws_size;
  const float* x     = (const float*)d_in[0];
  const float* W_in  = (const float*)d_in[1];
  const float* b_in  = (const float*)d_in[2];
  const float* W_mod = (const float*)d_in[3];
  const float* b_mod = (const float*)d_in[4];
  const float* Wr1   = (const float*)d_in[5];
  const float* br1   = (const float*)d_in[6];
  const float* Wr2   = (const float*)d_in[7];
  const float* br2   = (const float*)d_in[8];
  const float* Wc    = (const float*)d_in[9];
  const float* bc    = (const float*)d_in[10];
  const float* W_out = (const float*)d_in[11];
  const float* b_out = (const float*)d_in[12];
  float* out = (float*)d_out;

  char* w = (char*)d_ws;
  size_t off = 0;
  auto carve = [&](size_t bytes) -> void* {
    void* p = w + off;
    off += (bytes + 255) & ~(size_t)255;
    return p;
  };
  bf16_t* orig  = (bf16_t*)carve((size_t)Bn * Hn * 2);
  bf16_t* hA    = (bf16_t*)carve((size_t)Bn * Hn * 2);
  bf16_t* hB    = (bf16_t*)carve((size_t)Bn * Hn * 2);
  bf16_t* xm    = (bf16_t*)carve((size_t)Bn * Hn * 2);
  bf16_t* xbf   = (bf16_t*)carve((size_t)Bn * INn * 2);
  bf16_t* WtIn  = (bf16_t*)carve((size_t)INn * Hn * 2);
  bf16_t* WtMod = (bf16_t*)carve((size_t)Mn * Hn * Hn * 2);
  bf16_t* WtR1  = (bf16_t*)carve((size_t)Hn * Rn * 2);
  bf16_t* WtOut = (bf16_t*)carve((size_t)Hn * OUTn * 2);
  float*  rbuf  = (float*)carve((size_t)Bn * Rn * 4);
  float*  probs = (float*)carve((size_t)Bn * Mn * 4);

  dim3 blk(256);
  // one-time precision conversions (weights transposed to [N][K] bf16)
  conv_f32_to_bf16<<<(Bn * INn) / 256, blk, 0, stream>>>(x, xbf);
  convt_f32_to_bf16t<<<dim3((INn * Hn) / 256, 1), blk, 0, stream>>>(W_in, WtIn, INn, Hn);
  convt_f32_to_bf16t<<<dim3((Hn * Hn) / 256, Mn), blk, 0, stream>>>(W_mod, WtMod, Hn, Hn);
  convt_f32_to_bf16t<<<dim3((Hn * Rn) / 256, 1), blk, 0, stream>>>(Wr1, WtR1, Hn, Rn);
  convt_f32_to_bf16t<<<dim3((Hn * OUTn) / 256, 1), blk, 0, stream>>>(W_out, WtOut, Hn, OUTn);

  // h = x @ W_in + b_in  -> orig (bf16)
  gemm_bf16_kernel<false, true>
      <<<dim3(Hn / 128, Bn / 128), blk, 0, stream>>>(xbf, WtIn, b_in, orig, Hn, INn);

  const bf16_t* hcur = orig;
  for (int c = 0; c < CYC; ++c) {
    modulate_kernel<<<(Bn * Hn) / 256, blk, 0, stream>>>(orig, Wc, bc, xm, (float)c, Hn);
    gemm_bf16_kernel<true, false>
        <<<dim3(Rn / 128, Bn / 128), blk, 0, stream>>>(xm, WtR1, br1, rbuf, Rn, Hn);
    router_kernel<<<Bn / 256, blk, 0, stream>>>(rbuf, Wr2, br2, probs);
    bf16_t* hnext = (c & 1) ? hB : hA;
    expert_moe_kernel<<<dim3(Hn / 128, Bn / 128), blk, 0, stream>>>(
        hcur, WtMod, b_mod, probs, hnext, Hn);
    hcur = hnext;
  }

  // out = h @ W_out + b_out (f32)
  gemm_bf16_kernel<false, false>
      <<<dim3(OUTn / 128, Bn / 128), blk, 0, stream>>>(hcur, WtOut, b_out, out, OUTn, Hn);
}
